// Head_29351806500984
// MI455X (gfx1250) — matrix operand order repaired and checked by reading, not compiled, not run on hardware
//
#include <hip/hip_runtime.h>

#define BATCH 4
#define TT    2048
#define BT    (BATCH * TT)     // 8192 rows
#define NE    1024
#define HS    64
#define WPB   4                // waves per block

// softmax scale 1/sqrt(1024) with log2(e) folded in (exp2-based softmax)
#define QSCALE 0.0450842200f

typedef __attribute__((ext_vector_type(16))) _Float16 v16h;
typedef __attribute__((ext_vector_type(8)))  _Float16 v8h;
typedef __attribute__((ext_vector_type(8)))  float    v8f;
typedef __attribute__((ext_vector_type(4)))  float    v4f;

static __device__ __forceinline__ v8f wmma_f16(v16h a, v16h b, v8f c) {
  return __builtin_amdgcn_wmma_f32_16x16x32_f16(false, a, false, b, (short)0, c,
                                                false, false);
}

// B-operand: 16 contiguous f16 (lane-local 32B)
static __device__ __forceinline__ v16h load_b16(const _Float16* p) {
  union { v16h v; v8h h[2]; } u;
  u.h[0] = *(const v8h*)(p);
  u.h[1] = *(const v8h*)(p + 8);
  return u.v;
}

// A-operand: chunks at +0 and +8 (contiguous k = 16h+s, same order as load_b16)
static __device__ __forceinline__ v16h load_a16(const _Float16* p) {
  union { v16h v; v8h h[2]; } u;
  u.h[0] = *(const v8h*)(p);
  u.h[1] = *(const v8h*)(p + 8);
  return u.v;
}

// ---------------------------------------------------------------------------
// fp32 -> f16 streaming convert (8 elements/thread, n % 8 == 0)
// ---------------------------------------------------------------------------
__global__ __launch_bounds__(256)
void cvt_f16_kernel(const float* __restrict__ src, _Float16* __restrict__ dst,
                    int n)
{
  const int i = (blockIdx.x * 256 + threadIdx.x) * 8;
  if (i >= n) return;
  v4f a = *(const v4f*)(src + i);
  v4f b = *(const v4f*)(src + i + 4);
  v8h o;
#pragma unroll
  for (int j = 0; j < 4; ++j) { o[j] = (_Float16)a[j]; o[4 + j] = (_Float16)b[j]; }
  *(v8h*)(dst + i) = o;
}

// ---------------------------------------------------------------------------
// Plan A projection: all-f16, double-buffered LDS W stage, 4 live B operands.
// Block = 4 waves = 64 rows x 64 cols of one matrix; wave = 16 rows, 4 tiles.
// ---------------------------------------------------------------------------
__global__ __launch_bounds__(128)
void qkv_proj_f16(const _Float16* __restrict__ xh,   // [8192][1024]
                  const _Float16* __restrict__ Wh,   // [3][64][1024]
                  _Float16* __restrict__ Qo,
                  _Float16* __restrict__ Ko,
                  _Float16* __restrict__ Vt)
{
  __shared__ __align__(16) _Float16 Wl[2][64][40];   // double-buffered (2x5KB)

  const int tid  = threadIdx.x;
  const int lane = tid & 31;
  const int wv   = tid >> 5;
  const int m    = lane & 15;
  const int hi   = lane >> 4;
  const int row0 = blockIdx.x * 64 + wv * 16;
  const int mat  = blockIdx.y;

  const _Float16* W    = Wh + (size_t)mat * HS * NE;
  const _Float16* arow = xh + (size_t)(row0 + m) * NE + 16 * hi;

  // cooperative W staging: thread t copies 16 f16 of row n = t/2
  const int wn = tid >> 1;
  const int wh = (tid & 1) * 16;
  const _Float16* wsrc = W + (size_t)wn * NE + wh;

  // preload k = 0 chunk
  *(v8h*)(&Wl[0][wn][wh])     = *(const v8h*)(wsrc);
  *(v8h*)(&Wl[0][wn][wh + 8]) = *(const v8h*)(wsrc + 8);

  v8f c[4] = {};
  for (int k = 0; k < NE; k += 32) {
    const int buf = (k >> 5) & 1;
    __syncthreads();                         // stage of `buf` complete
    if (k + 32 < NE) {                       // prefetch next chunk
      *(v8h*)(&Wl[buf ^ 1][wn][wh])     = *(const v8h*)(wsrc + k + 32);
      *(v8h*)(&Wl[buf ^ 1][wn][wh + 8]) = *(const v8h*)(wsrc + k + 40);
    }
    const v16h a  = load_a16(arow + k);
    const v16h b0 = load_b16(&Wl[buf][ 0 + m][16 * hi]);
    const v16h b1 = load_b16(&Wl[buf][16 + m][16 * hi]);
    const v16h b2 = load_b16(&Wl[buf][32 + m][16 * hi]);
    const v16h b3 = load_b16(&Wl[buf][48 + m][16 * hi]);
    c[0] = wmma_f16(a, b0, c[0]);
    c[1] = wmma_f16(a, b1, c[1]);
    c[2] = wmma_f16(a, b2, c[2]);
    c[3] = wmma_f16(a, b3, c[3]);
  }

  if (mat == 2) {
#pragma unroll
    for (int ct = 0; ct < 4; ++ct) {
      v8h o;
#pragma unroll
      for (int r = 0; r < 8; ++r) o[r] = (_Float16)c[ct][r];
      *(v8h*)(Vt + (size_t)(ct * 16 + m) * BT + row0 + 8 * hi) = o;
    }
  } else {
    const float scl = (mat == 0) ? QSCALE : 1.0f;
    _Float16* dst = (mat == 0) ? Qo : Ko;
#pragma unroll
    for (int ct = 0; ct < 4; ++ct)
#pragma unroll
      for (int r = 0; r < 8; ++r)
        dst[(size_t)(row0 + r + 8 * hi) * HS + ct * 16 + m] =
            (_Float16)(c[ct][r] * scl);
  }
}

// ---------------------------------------------------------------------------
// Fallback projection (small workspace): converts fp32 operands on the fly.
// ---------------------------------------------------------------------------
__global__ __launch_bounds__(128)
void qkv_proj_f32(const float* __restrict__ x,
                  const float* __restrict__ Wq,
                  const float* __restrict__ Wk,
                  const float* __restrict__ Wv,
                  _Float16* __restrict__ Qo,
                  _Float16* __restrict__ Ko,
                  _Float16* __restrict__ Vt)
{
  const int lane = threadIdx.x & 31;
  const int wv   = threadIdx.x >> 5;
  const int m    = lane & 15;
  const int hi   = lane >> 4;
  const int row0 = blockIdx.x * 16;
  const int mat  = blockIdx.y;

  const float* W = (mat == 0) ? Wq : (mat == 1) ? Wk : Wv;
  const int n    = wv * 16 + m;

  const float* xrow = x + (size_t)(row0 + m) * NE + 16 * hi;
  const float* wrow = W + (size_t)n * NE + 16 * hi;

  v8f c = {};
  for (int k = 0; k < NE; k += 32) {
    v4f a0 = *(const v4f*)(xrow + k);
    v4f a1 = *(const v4f*)(xrow + k + 4);
    v4f a2 = *(const v4f*)(xrow + k + 8);
    v4f a3 = *(const v4f*)(xrow + k + 12);
    v4f b0 = *(const v4f*)(wrow + k);
    v4f b1 = *(const v4f*)(wrow + k + 4);
    v4f b2 = *(const v4f*)(wrow + k + 8);
    v4f b3 = *(const v4f*)(wrow + k + 12);
    v16h a, b;
#pragma unroll
    for (int i = 0; i < 4; ++i) {
      a[i] = (_Float16)a0[i]; a[i + 4] = (_Float16)a1[i];
      a[i + 8] = (_Float16)a2[i]; a[i + 12] = (_Float16)a3[i];
      b[i] = (_Float16)b0[i]; b[i + 4] = (_Float16)b1[i];
      b[i + 8] = (_Float16)b2[i]; b[i + 12] = (_Float16)b3[i];
    }
    c = wmma_f16(a, b, c);
  }

  if (mat == 2) {
    v8h o;
#pragma unroll
    for (int r = 0; r < 8; ++r) o[r] = (_Float16)c[r];
    *(v8h*)(Vt + (size_t)n * BT + row0 + 8 * hi) = o;
  } else {
    const float scl = (mat == 0) ? QSCALE : 1.0f;
    _Float16* dst = (mat == 0) ? Qo : Ko;
#pragma unroll
    for (int r = 0; r < 8; ++r)
      dst[(size_t)(row0 + r + 8 * hi) * HS + n] = (_Float16)(c[r] * scl);
  }
}

// ---------------------------------------------------------------------------
// Flash attention: 4 waves per 16-query tile; wave w takes key blocks
// j % 4 == w.  Row sums computed by an extra WMMA against a ones matrix
// (no cross-lane sum reduction); flash-combine across waves via LDS.
// ---------------------------------------------------------------------------
__global__ __launch_bounds__(32 * WPB)
void attn_kernel(const _Float16* __restrict__ Q,
                 const _Float16* __restrict__ K,
                 const _Float16* __restrict__ Vt,
                 float* __restrict__ out)
{
  __shared__ __align__(16) _Float16 plds[WPB][16][40];
  __shared__ float red_m[WPB][16];
  __shared__ float red_l[WPB][16];
  __shared__ float obuf[16][68];                    // padded vs. ds_add banks

  const int tid  = threadIdx.x;
  const int lane = tid & 31;
  const int wv   = tid >> 5;
  const int m    = lane & 15;
  const int hi   = lane >> 4;

  const int b  = blockIdx.x >> 7;                   // batch 0..3
  const int q0 = (blockIdx.x & 127) * 16;           // query tile in batch

  const _Float16* Qb = Q  + (size_t)b * TT * HS;
  const _Float16* Kb = K  + (size_t)b * TT * HS;
  const _Float16* Vb = Vt + (size_t)b * TT;
  float*          ob = out + (size_t)b * TT * HS;

  for (int i = tid; i < 16 * 68; i += 32 * WPB)     // zero combine buffer
    ((float*)obuf)[i] = 0.0f;

  const _Float16* qrow = Qb + (size_t)(q0 + m) * HS + 16 * hi;
  const v16h aq0 = load_a16(qrow);                  // pre-scaled by log2e/32
  const v16h aq1 = load_a16(qrow + 32);

  v16h ones;
#pragma unroll
  for (int i = 0; i < 16; ++i) ones[i] = (_Float16)1.0f;

  v8f acc0 = {}, acc1 = {}, acc2 = {}, acc3 = {}, lacc = {};
  float mrow[8];
#pragma unroll
  for (int r = 0; r < 8; ++r) mrow[r] = -__builtin_inff();

  const int jmax = (q0 + 15) >> 5;
  _Float16 (*P)[40] = plds[wv];

  for (int j = wv; j <= jmax; j += WPB) {
    const int k0 = j * 32;

    // ---- S = Q*K^T : 4 live B operands, back-to-back WMMAs ----
    const _Float16* kr0 = Kb + (size_t)(k0 + m) * HS + 16 * hi;
    const _Float16* kr1 = kr0 + 16 * HS;
    const v16h bk00 = load_b16(kr0);
    const v16h bk01 = load_b16(kr0 + 32);
    const v16h bk10 = load_b16(kr1);
    const v16h bk11 = load_b16(kr1 + 32);
    v8f s0 = {}, s1 = {};
    s0 = wmma_f16(aq0, bk00, s0);
    s0 = wmma_f16(aq1, bk01, s0);
    s1 = wmma_f16(aq0, bk10, s1);
    s1 = wmma_f16(aq1, bk11, s1);

    // issue V loads now; latency hidden under softmax VALU work
    const _Float16* vcol = Vb + (size_t)m * BT + k0 + 16 * hi;
    const v16h bv0 = load_b16(vcol);
    const v16h bv1 = load_b16(vcol + 16 * BT);
    const v16h bv2 = load_b16(vcol + 32 * BT);
    const v16h bv3 = load_b16(vcol + 48 * BT);

    if (j == jmax) {                                // diagonal block mask
#pragma unroll
      for (int r = 0; r < 8; ++r) {
        const int qg = q0 + r + 8 * hi;
        if (k0 + m > qg)      s0[r] = -__builtin_inff();
        if (k0 + 16 + m > qg) s1[r] = -__builtin_inff();
      }
    }

#pragma unroll
    for (int r = 0; r < 8; ++r) {
      float v = fmaxf(s0[r], s1[r]);
      v = fmaxf(v, __shfl_xor(v, 1, 32));
      v = fmaxf(v, __shfl_xor(v, 2, 32));
      v = fmaxf(v, __shfl_xor(v, 4, 32));
      v = fmaxf(v, __shfl_xor(v, 8, 32));
      const float mn = fmaxf(mrow[r], v);
      const float al = __builtin_amdgcn_exp2f(mrow[r] - mn);
      mrow[r] = mn;
      const float p0 = __builtin_amdgcn_exp2f(s0[r] - mn);
      const float p1 = __builtin_amdgcn_exp2f(s1[r] - mn);
      P[r + 8 * hi][m]      = (_Float16)p0;
      P[r + 8 * hi][16 + m] = (_Float16)p1;
      acc0[r] *= al; acc1[r] *= al; acc2[r] *= al; acc3[r] *= al;
      lacc[r] *= al;
    }

    asm volatile("s_wait_dscnt 0x0" ::: "memory");  // P stores -> A reload

    const v16h ap = load_a16(&P[m][16 * hi]);
    acc0 = wmma_f16(ap, bv0, acc0);
    acc1 = wmma_f16(ap, bv1, acc1);
    acc2 = wmma_f16(ap, bv2, acc2);
    acc3 = wmma_f16(ap, bv3, acc3);
    lacc = wmma_f16(ap, ones, lacc);                // row sums, no shuffles
  }

  // ---- flash combine across the 4 waves ----
  if (m == 0) {
#pragma unroll
    for (int r = 0; r < 8; ++r) red_m[wv][r + 8 * hi] = mrow[r];
  }
  __syncthreads();

  float fw[8];
#pragma unroll
  for (int r = 0; r < 8; ++r) {
    const int row = r + 8 * hi;
    const float ms = fmaxf(fmaxf(red_m[0][row], red_m[1][row]),
                           fmaxf(red_m[2][row], red_m[3][row]));
    fw[r] = __builtin_amdgcn_exp2f(mrow[r] - ms);   // 0 for idle waves
  }
  if (m == 0) {
#pragma unroll
    for (int r = 0; r < 8; ++r) red_l[wv][r + 8 * hi] = lacc[r] * fw[r];
  }
#pragma unroll
  for (int r = 0; r < 8; ++r) {
    const int row = r + 8 * hi;
    atomicAdd(&obuf[row][m],      acc0[r] * fw[r]);
    atomicAdd(&obuf[row][16 + m], acc1[r] * fw[r]);
    atomicAdd(&obuf[row][32 + m], acc2[r] * fw[r]);
    atomicAdd(&obuf[row][48 + m], acc3[r] * fw[r]);
  }
  __syncthreads();

  // wave wv writes h-tile wv
#pragma unroll
  for (int r = 0; r < 8; ++r) {
    const int row = r + 8 * hi;
    const float ls = red_l[0][row] + red_l[1][row] +
                     red_l[2][row] + red_l[3][row];
    ob[(size_t)(q0 + row) * HS + wv * 16 + m] = obuf[row][wv * 16 + m] / ls;
  }
}

extern "C" void kernel_launch(void* const* d_in, const int* in_sizes, int n_in,
                              void* d_out, int out_size, void* d_ws, size_t ws_size,
                              hipStream_t stream) {
  const float* x  = (const float*)d_in[0];
  const float* Wq = (const float*)d_in[1];
  const float* Wk = (const float*)d_in[2];
  const float* Wv = (const float*)d_in[3];

  _Float16* Qs = (_Float16*)d_ws;                 // [8192,64] (pre-scaled Q)
  _Float16* Ks = Qs + (size_t)BT * HS;
  _Float16* Vs = Ks + (size_t)BT * HS;            // [64,8192] transposed V

  const size_t base_bytes  = (size_t)3 * BT * HS * 2;               // 3 MB
  const size_t planA_bytes = base_bytes + (size_t)BT * NE * 2
                                        + (size_t)3 * HS * NE * 2;  // ~19.4 MB

  if (ws_size >= planA_bytes) {
    _Float16* xh = Vs + (size_t)BT * HS;          // [8192,1024] f16
    _Float16* Wh = xh + (size_t)BT * NE;          // [3][64,1024] f16
    cvt_f16_kernel<<<(BT * NE) / (256 * 8), 256, 0, stream>>>(x, xh, BT * NE);
    cvt_f16_kernel<<<(HS * NE) / (256 * 8), 256, 0, stream>>>(Wq, Wh, HS * NE);
    cvt_f16_kernel<<<(HS * NE) / (256 * 8), 256, 0, stream>>>(Wk, Wh + HS * NE,
                                                              HS * NE);
    cvt_f16_kernel<<<(HS * NE) / (256 * 8), 256, 0, stream>>>(Wv, Wh + 2 * HS * NE,
                                                              HS * NE);
    qkv_proj_f16<<<dim3(BT / 64, 3), 128, 0, stream>>>(xh, Wh, Qs, Ks, Vs);
  } else {
    qkv_proj_f32<<<dim3(BT / 16, 3), 128, 0, stream>>>(x, Wq, Wk, Wv, Qs, Ks, Vs);
  }

  attn_kernel<<<dim3(BT / 16), 32 * WPB, 0, stream>>>(Qs, Ks, Vs, (float*)d_out);
}
